// CNN_65867618451726
// MI455X (gfx1250) — matrix operand order repaired and checked by reading, not compiled, not run on hardware
//
#include <hip/hip_runtime.h>

typedef _Float16 half_t;
typedef __attribute__((ext_vector_type(16))) _Float16 v16h;
typedef __attribute__((ext_vector_type(8)))  _Float16 v8h;
typedef __attribute__((ext_vector_type(8)))  float    v8f;

#define BN_EPS 1e-5f
#define KSTEP  64

// stats layout per stage (256 floats): [0..63]=sum  [64..127]=sumsq  [128..191]=mean  [192..255]=inv_std

// ---------------- input-moment reduction (1 channel) ----------------
__global__ void reduce_moments_1c(const float* __restrict__ x, size_t n, float* __restrict__ stats) {
    __shared__ float ssum[256], ssq[256];
    float s = 0.f, q = 0.f;
    for (size_t i = (size_t)blockIdx.x * blockDim.x + threadIdx.x; i < n;
         i += (size_t)gridDim.x * blockDim.x) {
        float v = x[i]; s += v; q += v * v;
    }
    ssum[threadIdx.x] = s; ssq[threadIdx.x] = q;
    __syncthreads();
    for (int o = 128; o > 0; o >>= 1) {
        if ((int)threadIdx.x < o) {
            ssum[threadIdx.x] += ssum[threadIdx.x + o];
            ssq[threadIdx.x]  += ssq[threadIdx.x + o];
        }
        __syncthreads();
    }
    if (threadIdx.x == 0) { atomicAdd(&stats[0], ssum[0]); atomicAdd(&stats[64], ssq[0]); }
}

__global__ void finalize_stats(float* __restrict__ stats, int C, float invCnt) {
    int c = threadIdx.x;
    if (c < C) {
        float mean = stats[c] * invCnt;
        float var  = stats[64 + c] * invCnt - mean * mean;
        stats[128 + c] = mean;
        stats[192 + c] = rsqrtf(var + BN_EPS);
    }
}

// x (f32, [N,1,D,H,W]) -> BN-normalized f16 written into the interior of a zero-padded
// volume [N,1,Dp,Hp,Wp] (halo 3). grid: (z*H+y, n), block: W-covering threads.
__global__ void normalize_input_padded(const float* __restrict__ x, half_t* __restrict__ xp,
                                       const float* __restrict__ stats,
                                       int D, int H, int W, int Dp, int Hp, int Wp) {
    int n = blockIdx.y, row = blockIdx.x;
    int z = row / H, y = row - z * H;
    int xw = threadIdx.x;
    if (xw < W) {
        float mean = stats[128], inv = stats[192];
        float v = x[(((size_t)n * D + z) * H + y) * W + xw];
        xp[(((size_t)n * Dp + z + 3) * Hp + (y + 3)) * Wp + (xw + 3)] = (half_t)((v - mean) * inv);
    }
}

// compact conv output a ([N,C,D,H,W] f16) -> BN+bias+ReLU into interior of padded
// volume ap ([N,C,Dp,Hp,Wp], halo 3). grid: (z*H+y, n*C+c).
__global__ void norm_bias_relu_padded(const half_t* __restrict__ a, half_t* __restrict__ ap,
                                      const float* __restrict__ stats, const float* __restrict__ bias,
                                      int C, int D, int H, int W, int Dp, int Hp, int Wp) {
    int nc = blockIdx.y;
    int c  = nc % C;
    int row = blockIdx.x;
    int z = row / H, y = row - z * H;
    int xw = threadIdx.x;
    if (xw < W) {
        float v = (float)a[(((size_t)nc * D + z) * H + y) * W + xw];
        v = (v - stats[128 + c]) * stats[192 + c] + bias[c];
        v = v > 0.f ? v : 0.f;
        ap[(((size_t)nc * Dp + z + 3) * Hp + (y + 3)) * Wp + (xw + 3)] = (half_t)v;
    }
}

// f32 weights [Cout][Ktot] -> zero-padded f16 [CoutPad][Kpad]
__global__ void convert_weights(const float* __restrict__ w, half_t* __restrict__ dst,
                                int Cout, int CoutPad, int Ktot, int Kpad) {
    int i = blockIdx.x * blockDim.x + threadIdx.x;
    int tot = CoutPad * Kpad;
    if (i < tot) {
        int c = i / Kpad, k = i - c * Kpad;
        dst[i] = (c < Cout && k < Ktot) ? (half_t)w[c * Ktot + k] : (half_t)0.f;
    }
}

// k -> linear offset into the PADDED input volume: ((cin*Dp+kz)*Hp+ky)*Wp+kx.
// pad entries (gk >= Ktot) get offset 0: always a safe address, and their weights are zero.
__global__ void build_ktab(int* __restrict__ tab, int Ktot, int Kpad, int Dp, int Hp, int Wp) {
    int gk = blockIdx.x * blockDim.x + threadIdx.x;
    if (gk < Kpad) {
        int off = 0;
        if (gk < Ktot) {
            int cin = gk / 125;
            int rr  = gk - cin * 125;
            int kz  = rr / 25; rr -= kz * 25;
            int ky  = rr / 5;  int kx = rr - ky * 5;
            off = ((cin * Dp + kz) * Hp + ky) * Wp + kx;
        }
        tab[gk] = off;
    }
}

// epilogue helper: store one 16x8 accumulator octet + accumulate BN moments
__device__ __forceinline__ void store_tile(const v8f& acc, half_t* __restrict__ out,
                                           int g0, int M, int Sout, int Cout, int cgl,
                                           float& lsum, float& lsq) {
    if (g0 < M) {
        int n2 = g0 / Sout;
        int rem = g0 - n2 * Sout;
        #pragma unroll
        for (int r = 0; r < 8; ++r) {
            if (g0 + r < M) {
                float v = acc[r];
                out[((size_t)n2 * Cout + cgl) * (size_t)Sout + rem] = (half_t)v;
                lsum += v; lsq += v * v;
            }
            if (++rem == Sout) { rem = 0; ++n2; }
        }
    }
}

// ---------------- implicit-GEMM conv via v_wmma_f32_16x16x32_f16 ----------------
// Input is a 3D zero-padded f16 volume (halo 3 per face) -> the im2col gather is a
// pure pointer-chase: one table offset + one u16 load per element, no bounds checks.
// block = 128 threads = 4 waves; blockIdx.x = 32-row M-tile; wave w owns channels
// [16w,16w+16). 4 WMMAs per KSTEP=64 iteration per wave.
__global__ __launch_bounds__(128) void conv_wmma(
    const half_t* __restrict__ in, const half_t* __restrict__ wt,
    const int* __restrict__ ktab,
    half_t* __restrict__ out, float* __restrict__ stats,
    int N, int Cin, int Dp, int Hp, int Wp,
    int Cout, int CoutPad, int Do, int Ho, int Wo,
    int Kpad)
{
    __shared__ __align__(16) half_t a_tile[2][16 * KSTEP];

    const int HpWp = Hp * Wp;
    const int HWo  = Ho * Wo;
    const int Sout = Do * HWo;
    const int M    = N * Sout;

    const int tid  = threadIdx.x;
    const int wave = tid >> 5;
    const int lane = tid & 31;

    // gather assignment: row lm (0..15) of each sub-tile, k-octet [lk0, lk0+8)
    const int lm  = tid >> 3;
    const int lk0 = (tid & 7) * 8;
    int  basei[2];
    bool mv[2];
    #pragma unroll
    for (int t = 0; t < 2; ++t) {
        int gm = blockIdx.x * 32 + t * 16 + lm;
        mv[t] = (gm < M);
        basei[t] = 0;
        if (mv[t]) {
            int n = gm / Sout; int r = gm - n * Sout;
            int az = r / HWo;  r -= az * HWo;
            int ay = r / Wo;   int ax = r - ay * Wo;
            // (+3 halo) cancels the (-3) conv pad: start corner = (2az, 2ay, 2ax)
            basei[t] = (n * Cin * Dp + 2 * az) * HpWp + (2 * ay) * Wp + 2 * ax;
        }
    }

    const bool waveActive = (wave * 16) < CoutPad;       // wave-uniform
    const int  ncol = lane & 15;
    const int  hi   = lane >> 4;
    const int  cgl  = wave * 16 + ncol;
    const half_t* wrow = wt + (size_t)cgl * (size_t)Kpad;

    v8f acc0 = {};
    v8f acc1 = {};

    for (int kb = 0; kb < Kpad; kb += KSTEP) {
        // 8 consecutive tap offsets (merges to b128 loads, L0/L2 resident)
        int e[8];
        #pragma unroll
        for (int j = 0; j < 8; ++j) e[j] = ktab[kb + lk0 + j];

        #pragma unroll
        for (int t = 0; t < 2; ++t) {
            v8h u = {};
            if (mv[t]) {
                #pragma unroll
                for (int j = 0; j < 8; ++j) u[j] = in[(size_t)(basei[t] + e[j])];
            }
            *(v8h*)&a_tile[t][lm * KSTEP + lk0] = u;   // 16B aligned ds_store
        }
        __syncthreads();

        if (waveActive) {
            if (kb + 4 * KSTEP < Kpad) __builtin_prefetch(wrow + kb + 4 * KSTEP, 0, 0);
            v16h b0, b1, a00, a01, a10, a11;
            const int m = lane & 15;
            #pragma unroll
            for (int p = 0; p < 8; ++p) {
                // B (32x16 f16): lanes 0-15 hold K=0..15, lanes 16-31 hold K=16..31
                int kbx = hi * 16 + 2 * p;
                b0[2 * p]     = wrow[kb + kbx];
                b0[2 * p + 1] = wrow[kb + kbx + 1];
                b1[2 * p]     = wrow[kb + 32 + kbx];
                b1[2 * p + 1] = wrow[kb + 32 + kbx + 1];
                // A (16x32 f16): lanes 0-15 hold K=0..15, lanes 16-31 hold K=16..31
                int ka = hi * 16 + 2 * p;
                a00[2 * p]     = a_tile[0][m * KSTEP + ka];
                a00[2 * p + 1] = a_tile[0][m * KSTEP + ka + 1];
                a01[2 * p]     = a_tile[0][m * KSTEP + 32 + ka];
                a01[2 * p + 1] = a_tile[0][m * KSTEP + 32 + ka + 1];
                a10[2 * p]     = a_tile[1][m * KSTEP + ka];
                a10[2 * p + 1] = a_tile[1][m * KSTEP + ka + 1];
                a11[2 * p]     = a_tile[1][m * KSTEP + 32 + ka];
                a11[2 * p + 1] = a_tile[1][m * KSTEP + 32 + ka + 1];
            }
            // interleave accumulators to avoid back-to-back RAW on the same D
            acc0 = __builtin_amdgcn_wmma_f32_16x16x32_f16(false, a00, false, b0, (short)0, acc0, false, false);
            acc1 = __builtin_amdgcn_wmma_f32_16x16x32_f16(false, a10, false, b0, (short)0, acc1, false, false);
            acc0 = __builtin_amdgcn_wmma_f32_16x16x32_f16(false, a01, false, b1, (short)0, acc0, false, false);
            acc1 = __builtin_amdgcn_wmma_f32_16x16x32_f16(false, a11, false, b1, (short)0, acc1, false, false);
        }
        __syncthreads();
    }

    if (waveActive) {
        const bool cvalid = (cgl < Cout);
        float lsum = 0.f, lsq = 0.f;
        if (cvalid) {
            store_tile(acc0, out, blockIdx.x * 32 +      hi * 8, M, Sout, Cout, cgl, lsum, lsq);
            store_tile(acc1, out, blockIdx.x * 32 + 16 + hi * 8, M, Sout, Cout, cgl, lsum, lsq);
        }
        if (stats != nullptr && cvalid) {
            atomicAdd(&stats[cgl], lsum);
            atomicAdd(&stats[64 + cgl], lsq);
        }
    }
}

// global average pool: block per (n,c), S = 216
__global__ void pool_kernel(const half_t* __restrict__ h, float* __restrict__ pooled) {
    __shared__ float red[64];
    int b = blockIdx.x;
    float s = 0.f;
    for (int i = threadIdx.x; i < 216; i += 64) s += (float)h[(size_t)b * 216 + i];
    red[threadIdx.x] = s;
    __syncthreads();
    for (int o = 32; o > 0; o >>= 1) {
        if ((int)threadIdx.x < o) red[threadIdx.x] += red[threadIdx.x + o];
        __syncthreads();
    }
    if (threadIdx.x == 0) pooled[b] = red[0] * (1.f / 216.f);
}

// per-class batch BN with affine (gamma, beta); pooled layout [32][10]
__global__ void final_bn(const float* __restrict__ pooled, const float* __restrict__ gamma,
                         const float* __restrict__ beta, float* __restrict__ out) {
    __shared__ float vals[320];
    __shared__ float mean[10], inv[10];
    int t = threadIdx.x;
    if (t < 320) vals[t] = pooled[t];
    __syncthreads();
    if (t < 10) {
        float s = 0.f, q = 0.f;
        for (int n = 0; n < 32; ++n) { float v = vals[n * 10 + t]; s += v; q += v * v; }
        float m = s * (1.f / 32.f);
        mean[t] = m;
        inv[t]  = rsqrtf(q * (1.f / 32.f) - m * m + BN_EPS);
    }
    __syncthreads();
    if (t < 320) {
        int c = t % 10;
        out[t] = (vals[t] - mean[c]) * inv[c] * gamma[c] + beta[c];
    }
}

extern "C" void kernel_launch(void* const* d_in, const int* in_sizes, int n_in,
                              void* d_out, int out_size, void* d_ws, size_t ws_size,
                              hipStream_t stream) {
    (void)in_sizes; (void)n_in; (void)out_size; (void)ws_size;
    const float* x  = (const float*)d_in[0];
    const float* w0 = (const float*)d_in[1];
    const float* b0 = (const float*)d_in[2];
    const float* w1 = (const float*)d_in[3];
    const float* b1 = (const float*)d_in[4];
    const float* w2 = (const float*)d_in[5];
    const float* b2 = (const float*)d_in[6];
    const float* w3 = (const float*)d_in[7];
    const float* gamma = (const float*)d_in[8];
    const float* beta  = (const float*)d_in[9];
    float* out = (float*)d_out;

    // ---- sizes (bytes) ----
    const size_t XP_B  = (size_t)32 * 1  * 70 * 70 * 70 * 2;   // 21,952,000
    const size_t A0C_B = (size_t)32 * 52 * 33 * 33 * 33 * 2;   // 119,598,336
    const size_t A0P_B = (size_t)32 * 52 * 39 * 39 * 39 * 2;   // 197,413,632
    const size_t A1P_B = (size_t)32 * 52 * 24 * 24 * 24 * 2;   //  46,006,272
    const size_t A2P_B = (size_t)32 * 52 * 16 * 16 * 16 * 2;   //  13,631,488
    const size_t A3_B  = (size_t)32 * 10 * 216 * 2;

    // deterministic workspace carve-up (256B aligned)
    char* ws = (char*)d_ws;
    size_t off = 0;
    auto take = [&](size_t bytes) -> char* {
        char* p = ws + off;
        off = (off + bytes + 255) & ~(size_t)255;
        return p;
    };
    float*  stats  = (float*)take(4 * 256 * sizeof(float));   // 4 BN stages
    float*  pooled = (float*)take(320 * sizeof(float));
    char*   Rxp    = take(XP_B);     // holds: xp -> a1 (compact) -> a2 (compact)
    char*   Ra0    = take(A0C_B);    // holds: a0 (compact) -> {a1p @0, a2p @64MB}
    half_t* a0p    = (half_t*)take(A0P_B);
    half_t* a3     = (half_t*)take(A3_B);
    half_t* w0p = (half_t*)take((size_t)64 * 128  * 2);
    half_t* w1p = (half_t*)take((size_t)64 * 6528 * 2);
    half_t* w2p = (half_t*)take((size_t)64 * 6528 * 2);
    half_t* w3p = (half_t*)take((size_t)16 * 6528 * 2);
    int*    kt0 = (int*)take((size_t)128  * sizeof(int));
    int*    kt1 = (int*)take((size_t)6528 * sizeof(int));
    int*    kt2 = (int*)take((size_t)6528 * sizeof(int));
    int*    kt3 = (int*)take((size_t)6528 * sizeof(int));

    half_t* xp  = (half_t*)Rxp;
    half_t* a1c = (half_t*)Rxp;                      // 19.4MB <= XP_B, xp dead by then
    half_t* a2c = (half_t*)Rxp;                      // 3.3MB, a1c dead by then
    half_t* a0c = (half_t*)Ra0;
    half_t* a1p = (half_t*)Ra0;                      // 46MB, a0c dead by then
    half_t* a2p = (half_t*)(Ra0 + 67108864);         // disjoint from a1p, fits in region

    hipMemsetAsync(stats, 0, 4 * 256 * sizeof(float), stream);
    hipMemsetAsync(xp, 0, XP_B, stream);             // zero halos of padded input
    hipMemsetAsync(a0p, 0, A0P_B, stream);           // zero halos (dedicated region)

    // input BN -> padded f16
    reduce_moments_1c<<<512, 256, 0, stream>>>(x, (size_t)8388608, stats);
    finalize_stats<<<1, 64, 0, stream>>>(stats, 1, 1.f / 8388608.f);
    normalize_input_padded<<<dim3(64 * 64, 32), 64, 0, stream>>>(x, xp, stats,
                                                                 64, 64, 64, 70, 70, 70);

    // weights -> padded f16 ; tap-offset tables (into each layer's padded volume)
    convert_weights<<<(64 * 128  + 255) / 256, 256, 0, stream>>>(w0, w0p, 52, 64, 125,  128);
    convert_weights<<<(64 * 6528 + 255) / 256, 256, 0, stream>>>(w1, w1p, 52, 64, 6500, 6528);
    convert_weights<<<(64 * 6528 + 255) / 256, 256, 0, stream>>>(w2, w2p, 52, 64, 6500, 6528);
    convert_weights<<<(16 * 6528 + 255) / 256, 256, 0, stream>>>(w3, w3p, 10, 16, 6500, 6528);
    build_ktab<<<1, 128, 0, stream>>>(kt0, 125, 128, 70, 70, 70);
    build_ktab<<<26, 256, 0, stream>>>(kt1, 6500, 6528, 39, 39, 39);
    build_ktab<<<26, 256, 0, stream>>>(kt2, 6500, 6528, 24, 24, 24);
    build_ktab<<<26, 256, 0, stream>>>(kt3, 6500, 6528, 16, 16, 16);

    // layer 0: 70^3(p) -> 33^3, Cin=1   (M = 1149984 -> 35937 blocks)
    conv_wmma<<<35937, 128, 0, stream>>>(xp, w0p, kt0, a0c, stats + 256,
                                         32, 1, 70, 70, 70, 52, 64, 33, 33, 33, 128);
    finalize_stats<<<1, 64, 0, stream>>>(stats + 256, 52, 1.f / 1149984.f);
    norm_bias_relu_padded<<<dim3(33 * 33, 1664), 64, 0, stream>>>(a0c, a0p, stats + 256, b0,
                                                                  52, 33, 33, 33, 39, 39, 39);
    // a0c is dead now -> zero the halos of the aliased padded buffers in its region
    hipMemsetAsync(a1p, 0, A1P_B, stream);
    hipMemsetAsync(a2p, 0, A2P_B, stream);

    // layer 1: 39^3(p) -> 18^3, Cin=52  (M = 186624 -> 5832 blocks)  [dominant layer]
    conv_wmma<<<5832, 128, 0, stream>>>(a0p, w1p, kt1, a1c, stats + 512,
                                        32, 52, 39, 39, 39, 52, 64, 18, 18, 18, 6528);
    finalize_stats<<<1, 64, 0, stream>>>(stats + 512, 52, 1.f / 186624.f);
    norm_bias_relu_padded<<<dim3(18 * 18, 1664), 64, 0, stream>>>(a1c, a1p, stats + 512, b1,
                                                                  52, 18, 18, 18, 24, 24, 24);

    // layer 2: 24^3(p) -> 10^3          (M = 32000 -> 1000 blocks)
    conv_wmma<<<1000, 128, 0, stream>>>(a1p, w2p, kt2, a2c, stats + 768,
                                        32, 52, 24, 24, 24, 52, 64, 10, 10, 10, 6528);
    finalize_stats<<<1, 64, 0, stream>>>(stats + 768, 52, 1.f / 32000.f);
    norm_bias_relu_padded<<<dim3(10 * 10, 1664), 64, 0, stream>>>(a2c, a2p, stats + 768, b2,
                                                                  52, 10, 10, 10, 16, 16, 16);

    // layer 3: 16^3(p) -> 6^3, Cout=10  (M = 6912 -> 216 blocks), no BN stats
    conv_wmma<<<216, 128, 0, stream>>>(a2p, w3p, kt3, a3, (float*)nullptr,
                                       32, 52, 16, 16, 16, 10, 16, 6, 6, 6, 6528);

    pool_kernel<<<320, 64, 0, stream>>>(a3, pooled);
    final_bn<<<1, 320, 0, stream>>>(pooled, gamma, beta, out);
}
